// SOL1d_3332894622155
// MI455X (gfx1250) — matrix operand order repaired and checked by reading, not compiled, not run on hardware
//
#include <hip/hip_runtime.h>
#include <math.h>

typedef __bf16 bf16;
typedef __attribute__((ext_vector_type(8)))  __bf16 v8bf;
typedef __attribute__((ext_vector_type(16))) __bf16 v16bf;
typedef __attribute__((ext_vector_type(8)))  float  v8f;

#define NPTS   4096
#define BATCH  16
#define CH     64
#define MODES_ 2048
#define KFWD   2052          // modes + bw - 1
#define NFWD   2112          // KFWD padded up to multiple of 64

#define WMMA_BF16(a,b,c) \
  __builtin_amdgcn_wmma_f32_16x16x32_bf16(false,(a),false,(b),(short)0,(c),false,false)

__device__ __forceinline__ float gelu_exact(float x) {
  return 0.5f * x * (1.0f + erff(x * 0.70710678118654752f));
}

__device__ __forceinline__ void split_bf16(float v, bf16& h, bf16& l) {
  h = (bf16)v;
  l = (bf16)(v - (float)h);
}

// Load a v16bf fragment as two contiguous 16B pieces from LDS.
__device__ __forceinline__ v16bf frag16(const bf16* p0, const bf16* p1) {
  union { v8bf h[2]; v16bf v; } u;
  u.h[0] = *(const v8bf*)p0;
  u.h[1] = *(const v8bf*)p1;
  return u.v;
}

// ---------------------------------------------------------------------------
// DCT table generation. Forward table: fwd[k][n] = Tm[k,n], k<KFWD (zero-padded
// to NFWD rows).  Inverse table (transposed): inv[n][k] = Tm[k,n], k<2048.
// Stored as bf16 hi/lo splits.
// ---------------------------------------------------------------------------
__global__ void k_gen_fwd(bf16* __restrict__ hi, bf16* __restrict__ lo) {
  const int n = blockIdx.x * 256 + threadIdx.x;   // 0..4095
  const int k = blockIdx.y;                       // 0..NFWD-1
  float f = 0.0f;
  if (k < KFWD) {
    double v = cos(3.14159265358979323846 * (n + 0.5) * (double)k / (double)NPTS)
             * 0.022097086912079608;              // sqrt(2/4096)
    if (k == 0) v *= 0.70710678118654752;
    f = (float)v;
  }
  bf16 h, l; split_bf16(f, h, l);
  const size_t idx = (size_t)k * NPTS + n;
  hi[idx] = h; lo[idx] = l;
}

__global__ void k_gen_inv(bf16* __restrict__ hi, bf16* __restrict__ lo) {
  const int k = blockIdx.x * 256 + threadIdx.x;   // 0..2047
  const int n = blockIdx.y;                       // 0..4095
  double v = cos(3.14159265358979323846 * (n + 0.5) * (double)k / (double)NPTS)
           * 0.022097086912079608;
  if (k == 0) v *= 0.70710678118654752;
  const float f = (float)v;
  bf16 h, l; split_bf16(f, h, l);
  const size_t idx = (size_t)n * MODES_ + k;
  hi[idx] = h; lo[idx] = l;
}

// ---------------------------------------------------------------------------
// Generic guarded GEMM  C[M x N] = A[M x K] * B[K x N]  with bf16x3 WMMA.
// A: row-major hi/lo [M][K].  B: column-major hi/lo [N][K] (i.e. B^T rows).
// Output row remap: orow = (m / rpi) * rpo + roff + (m % rpi).
// Block: 128 threads = 4 waves, 64x64 tile, K step 32.
// ---------------------------------------------------------------------------
__global__ __launch_bounds__(128) void k_gemm(
    const bf16* __restrict__ Ahi, const bf16* __restrict__ Alo,
    const bf16* __restrict__ Bhi, const bf16* __restrict__ Blo,
    float* __restrict__ C,
    int M, int N, int K, int ldc, int rpi, int rpo, int roff)
{
  __shared__ bf16 sAh[64][32];
  __shared__ bf16 sAl[64][32];
  __shared__ bf16 sBh[64][32];
  __shared__ bf16 sBl[64][32];

  const int tid  = threadIdx.x;
  const int wave = tid >> 5;
  const int lane = tid & 31;
  const int m0 = blockIdx.y * 64;
  const int n0 = blockIdx.x * 64;

  v8f acc[4] = {};

  const int aRow = (wave << 4) + (lane & 15);     // wave's A row within tile
  const int ka   = (lane & 16) ? 8 : 0;           // A K-offset per ISA layout
  const int kb   = (lane & 16) ? 8 : 0;           // B K-offset per ISA layout

  for (int k0 = 0; k0 < K; k0 += 32) {
    // ---- stage tiles into LDS (guarded, zero-filled out of range) ----
    #pragma unroll
    for (int half = 0; half < 2; ++half) {
      const int g  = tid + half * 128;            // 0..255
      const int r  = g >> 2;                      // row 0..63
      const int c8 = (g & 3) << 3;                // col 0,8,16,24
      uint4 z = {0u, 0u, 0u, 0u};
      uint4 va_h = z, va_l = z, vb_h = z, vb_l = z;
      if (m0 + r < M) {
        const size_t base = (size_t)(m0 + r) * K + k0 + c8;
        va_h = *(const uint4*)(Ahi + base);
        va_l = *(const uint4*)(Alo + base);
      }
      if (n0 + r < N) {
        const size_t base = (size_t)(n0 + r) * K + k0 + c8;
        vb_h = *(const uint4*)(Bhi + base);
        vb_l = *(const uint4*)(Blo + base);
      }
      *(uint4*)&sAh[r][c8] = va_h;
      *(uint4*)&sAl[r][c8] = va_l;
      *(uint4*)&sBh[r][c8] = vb_h;
      *(uint4*)&sBl[r][c8] = vb_l;
    }
    __syncthreads();

    // ---- A fragments for this wave ----
    const v16bf ah = frag16(&sAh[aRow][ka], &sAh[aRow][ka + 16]);
    const v16bf al = frag16(&sAl[aRow][ka], &sAl[aRow][ka + 16]);

    // ---- 4 N-tiles, 3 WMMAs each (hi*hi + hi*lo + lo*hi) ----
    #pragma unroll
    for (int nt = 0; nt < 4; ++nt) {
      const int bCol = nt * 16 + (lane & 15);
      const v16bf bh = frag16(&sBh[bCol][kb], &sBh[bCol][kb + 16]);
      const v16bf bl = frag16(&sBl[bCol][kb], &sBl[bCol][kb + 16]);
      acc[nt] = WMMA_BF16(ah, bh, acc[nt]);
      acc[nt] = WMMA_BF16(ah, bl, acc[nt]);
      acc[nt] = WMMA_BF16(al, bh, acc[nt]);
    }
    __syncthreads();
  }

  // ---- store (C/D layout: lane = N col, VGPR j = M row (+8 for hi lanes)) ----
  #pragma unroll
  for (int nt = 0; nt < 4; ++nt) {
    const int n = n0 + nt * 16 + (lane & 15);
    if (n < N) {
      #pragma unroll
      for (int j = 0; j < 8; ++j) {
        const int m = m0 + (wave << 4) + ((lane >> 4) << 3) + j;
        if (m < M) {
          const int orow = (m / rpi) * rpo + roff + (m % rpi);
          C[(size_t)orow * ldc + n] = acc[nt][j];
        }
      }
    }
  }
}

// ---------------------------------------------------------------------------
// Prepare first-layer input: U rows 0..1 per batch = x transposed, plus
// bf16 hi/lo A-matrix Xt[(b*2+c)][n] for the lifting forward DCT.
// ---------------------------------------------------------------------------
__global__ void k_prep(const float* __restrict__ x, float* __restrict__ U,
                       bf16* __restrict__ Xh, bf16* __restrict__ Xl) {
  const int n = blockIdx.x * 256 + threadIdx.x;
  const int b = blockIdx.y;
  #pragma unroll
  for (int c = 0; c < 2; ++c) {
    const float v = x[((size_t)b * NPTS + n) * 2 + c];
    U[(size_t)(b * CH + c) * NPTS + n] = v;
    bf16 h, l; split_bf16(v, h, l);
    const size_t r = (size_t)(b * 2 + c) * NPTS + n;
    Xh[r] = h; Xl[r] = l;
  }
}

// ---------------------------------------------------------------------------
// Per-mode windowed channel mix:
//   O[b,o,x] = sum_{i<CI,w<5} Bspec[(b*CI+i)][x+w] * W[x,i,o,w]
// One block per mode x; spectral slice cached in LDS; weights read once and
// reused over 16 batches.  Emits bf16 hi/lo (A-matrix of the inverse GEMM).
// ---------------------------------------------------------------------------
__global__ __launch_bounds__(256) void k_window(
    const float* __restrict__ Bs, int ldb,
    const float* __restrict__ W,
    bf16* __restrict__ Oh, bf16* __restrict__ Ol,
    int CI, int CO)
{
  __shared__ float sB[BATCH * CH * 5];
  const int x   = blockIdx.x;         // 0..2047
  const int tid = threadIdx.x;
  const int tot = BATCH * CI * 5;
  for (int e = tid; e < tot; e += 256) {
    const int b = e / (CI * 5);
    const int r = e - b * CI * 5;
    const int i = r / 5, w = r - i * 5;
    sB[e] = Bs[(size_t)(b * CI + i) * ldb + x + w];
  }
  __syncthreads();

  const int o  = tid & 63;
  const int bq = tid >> 6;            // 0..3, handles b = bq, bq+4, bq+8, bq+12
  if (o < CO) {
    float acc[4] = {0.f, 0.f, 0.f, 0.f};
    for (int i = 0; i < CI; ++i) {
      #pragma unroll
      for (int w = 0; w < 5; ++w) {
        const float wt = W[(((size_t)x * CI + i) * CO + o) * 5 + w];
        #pragma unroll
        for (int jb = 0; jb < 4; ++jb) {
          acc[jb] += wt * sB[((bq + jb * 4) * CI + i) * 5 + w];
        }
      }
    }
    #pragma unroll
    for (int jb = 0; jb < 4; ++jb) {
      const int b = bq + jb * 4;
      const size_t idx = (size_t)(b * CO + o) * MODES_ + x;
      bf16 h, l; split_bf16(acc[jb], h, l);
      Oh[idx] = h; Ol[idx] = l;
    }
  }
}

// ---------------------------------------------------------------------------
// Pointwise channel mix: P[b,o,n] = pb[o] + sum_c pw[o,c] * U[b,c,n]
// ---------------------------------------------------------------------------
__global__ __launch_bounds__(256) void k_pw(
    const float* __restrict__ U, const float* __restrict__ pw,
    const float* __restrict__ pb, float* __restrict__ P)
{
  __shared__ float sU[CH][256];
  const int tid = threadIdx.x;
  const int n0  = blockIdx.x * 256;
  const int b   = blockIdx.y;
  for (int c = 0; c < CH; ++c)
    sU[c][tid] = U[(size_t)(b * CH + c) * NPTS + n0 + tid];
  __syncthreads();
  for (int o = 0; o < CH; ++o) {
    float acc = pb[o];
    #pragma unroll 8
    for (int c = 0; c < CH; ++c) acc += pw[o * CH + c] * sU[c][tid];
    P[(size_t)(b * CH + o) * NPTS + n0 + tid] = acc;
  }
}

// In-place gelu on U rows 2..63 (first-layer lifting output).
__global__ void k_gelu_first(float* __restrict__ U) {
  const int n = blockIdx.x * 256 + threadIdx.x;
  const int m = blockIdx.y;                         // 0..991
  const int b = m / 62, c = 2 + (m - b * 62);
  const size_t idx = (size_t)(b * CH + c) * NPTS + n;
  U[idx] = gelu_exact(U[idx]);
}

// Split U (f32) into bf16 hi/lo.
__global__ void k_split(const float* __restrict__ U,
                        bf16* __restrict__ Uh, bf16* __restrict__ Ul) {
  const size_t i = (size_t)blockIdx.x * 256 + threadIdx.x;
  bf16 h, l; split_bf16(U[i], h, l);
  Uh[i] = h; Ul[i] = l;
}

// Residual update U = U + gelu(P + S); also refresh bf16 hi/lo split of U.
__global__ void k_fuse(float* __restrict__ U, const float* __restrict__ P,
                       const float* __restrict__ S,
                       bf16* __restrict__ Uh, bf16* __restrict__ Ul) {
  const size_t i = (size_t)blockIdx.x * 256 + threadIdx.x;
  const float v = U[i] + gelu_exact(P[i] + S[i]);
  U[i] = v;
  bf16 h, l; split_bf16(v, h, l);
  Uh[i] = h; Ul[i] = l;
}

// Fused MLP head: out[b,n] = f2b + sum_j f2w[j]*gelu(f1b[j] + sum_c f1w[j,c]*U[b,c,n])
// (The reference's trailing Tm^T Tm round-trip is the identity: Tm is orthonormal.)
__global__ __launch_bounds__(256) void k_head(
    const float* __restrict__ U, const float* __restrict__ f1w,
    const float* __restrict__ f1b, const float* __restrict__ f2w,
    const float* __restrict__ f2b, float* __restrict__ out)
{
  __shared__ float sU[CH][256];
  const int tid = threadIdx.x;
  const int n0  = blockIdx.x * 256;
  const int b   = blockIdx.y;
  for (int c = 0; c < CH; ++c)
    sU[c][tid] = U[(size_t)(b * CH + c) * NPTS + n0 + tid];
  __syncthreads();
  float acc = f2b[0];
  for (int j = 0; j < 128; ++j) {
    float t = f1b[j];
    #pragma unroll 8
    for (int c = 0; c < CH; ++c) t += f1w[j * CH + c] * sU[c][tid];
    acc += f2w[j] * gelu_exact(t);
  }
  out[(size_t)b * NPTS + n0 + tid] = acc;
}

// ---------------------------------------------------------------------------
extern "C" void kernel_launch(void* const* d_in, const int* in_sizes, int n_in,
                              void* d_out, int out_size, void* d_ws, size_t ws_size,
                              hipStream_t stream)
{
  (void)in_sizes; (void)n_in; (void)out_size; (void)ws_size;

  const float* x   = (const float*)d_in[0];
  const float* Wl  = (const float*)d_in[1];
  const float* Wm[4]   = {(const float*)d_in[2],  (const float*)d_in[3],
                          (const float*)d_in[4],  (const float*)d_in[5]};
  const float* pw_w[4] = {(const float*)d_in[6],  (const float*)d_in[8],
                          (const float*)d_in[10], (const float*)d_in[12]};
  const float* pw_b[4] = {(const float*)d_in[7],  (const float*)d_in[9],
                          (const float*)d_in[11], (const float*)d_in[13]};
  const float* f1w = (const float*)d_in[14];
  const float* f1b = (const float*)d_in[15];
  const float* f2w = (const float*)d_in[16];
  const float* f2b = (const float*)d_in[17];

  char*  ws  = (char*)d_ws;
  size_t off = 0;
  auto take = [&](size_t bytes) -> char* {
    char* p = ws + off;
    off = (off + bytes + 255) & ~(size_t)255;
    return p;
  };

  bf16* fwdH = (bf16*)take((size_t)NFWD * NPTS * 2);
  bf16* fwdL = (bf16*)take((size_t)NFWD * NPTS * 2);
  bf16* invH = (bf16*)take((size_t)NPTS * MODES_ * 2);
  bf16* invL = (bf16*)take((size_t)NPTS * MODES_ * 2);
  float* U   = (float*)take((size_t)BATCH * CH * NPTS * 4);
  bf16*  Uh  = (bf16*)take((size_t)BATCH * CH * NPTS * 2);
  bf16*  Ul  = (bf16*)take((size_t)BATCH * CH * NPTS * 2);
  bf16*  Xh  = (bf16*)take((size_t)BATCH * 2 * NPTS * 2);
  bf16*  Xl  = (bf16*)take((size_t)BATCH * 2 * NPTS * 2);
  float* Bspec = (float*)take((size_t)BATCH * CH * NFWD * 4);
  bf16*  Oh  = (bf16*)take((size_t)BATCH * CH * MODES_ * 2);
  bf16*  Ol  = (bf16*)take((size_t)BATCH * CH * MODES_ * 2);
  float* P   = (float*)take((size_t)BATCH * CH * NPTS * 4);
  float* S   = (float*)take((size_t)BATCH * CH * NPTS * 4);

  const dim3 blk128(128), blk256(256);

  // DCT tables (every call; no cross-call caching allowed).
  k_gen_fwd<<<dim3(NPTS / 256, NFWD), blk256, 0, stream>>>(fwdH, fwdL);
  k_gen_inv<<<dim3(MODES_ / 256, NPTS), blk256, 0, stream>>>(invH, invL);

  // Lifting layer: channels 0..1 copied, 2..63 = gelu(pseudo(u, Wl)).
  k_prep<<<dim3(NPTS / 256, BATCH), blk256, 0, stream>>>(x, U, Xh, Xl);
  k_gemm<<<dim3(NFWD / 64, 1), blk128, 0, stream>>>(
      Xh, Xl, fwdH, fwdL, Bspec, BATCH * 2, NFWD, NPTS, NFWD, BATCH * 2, 0, 0);
  k_window<<<MODES_, blk256, 0, stream>>>(Bspec, NFWD, Wl, Oh, Ol, 2, 62);
  k_gemm<<<dim3(NPTS / 64, (BATCH * 62 + 63) / 64), blk128, 0, stream>>>(
      Oh, Ol, invH, invL, U, BATCH * 62, NPTS, MODES_, NPTS, 62, CH, 2);
  k_gelu_first<<<dim3(NPTS / 256, BATCH * 62), blk256, 0, stream>>>(U);
  k_split<<<dim3((BATCH * CH * NPTS) / 256), blk256, 0, stream>>>(U, Uh, Ul);

  // 4 spectral layers.
  for (int L = 0; L < 4; ++L) {
    k_gemm<<<dim3(NFWD / 64, (BATCH * CH) / 64), blk128, 0, stream>>>(
        Uh, Ul, fwdH, fwdL, Bspec, BATCH * CH, NFWD, NPTS, NFWD, BATCH * CH, 0, 0);
    k_window<<<MODES_, blk256, 0, stream>>>(Bspec, NFWD, Wm[L], Oh, Ol, CH, CH);
    k_pw<<<dim3(NPTS / 256, BATCH), blk256, 0, stream>>>(U, pw_w[L], pw_b[L], P);
    k_gemm<<<dim3(NPTS / 64, (BATCH * CH) / 64), blk128, 0, stream>>>(
        Oh, Ol, invH, invL, S, BATCH * CH, NPTS, MODES_, NPTS, BATCH * CH, 0, 0);
    k_fuse<<<dim3((BATCH * CH * NPTS) / 256), blk256, 0, stream>>>(U, P, S, Uh, Ul);
  }

  // MLP head (trailing DCT round-trip is identity; skipped).
  k_head<<<dim3(NPTS / 256, BATCH), blk256, 0, stream>>>(
      U, f1w, f1b, f2w, f2b, (float*)d_out);
}
